// NODE_5050881540422
// MI455X (gfx1250) — matrix operand order repaired and checked by reading, not compiled, not run on hardware
//
#include <hip/hip_runtime.h>
#include <hip/hip_bf16.h>

#define BATCH   65536
#define UNITS   128
#define IDIM    64
#define WPITCH  192
#define UNFOLDS 6
#define DT      0.1f
#define MTILE   64

typedef __attribute__((ext_vector_type(16))) _Float16 v16h;
typedef __attribute__((ext_vector_type(8)))  _Float16 v8h;
typedef __attribute__((ext_vector_type(2)))  _Float16 v2h;
typedef __attribute__((ext_vector_type(8)))  float    v8f;

__device__ __forceinline__ v8f wmma_f16(v16h a, v16h b, v8f c) {
  // 8 args: (neg_a, A, neg_b, B, c_mod, C, reuse_a, reuse_b)
  return __builtin_amdgcn_wmma_f32_16x16x32_f16(false, a, false, b, (short)0, c,
                                                false, false);
}

__device__ __forceinline__ float fast_tanh(float x) {
  // tanh(x) = 1 - 2/(exp(2x)+1) ; v_exp_f32 + v_rcp_f32 (TRANS ops co-exec with WMMA)
  float e = __expf(2.0f * x);
  return 1.0f - 2.0f * __builtin_amdgcn_rcpf(e + 1.0f);
}

__device__ __forceinline__ v16h cat16(v8h lo, v8h hi) {
  return __builtin_shufflevector(lo, hi, 0, 1, 2, 3, 4, 5, 6, 7,
                                         8, 9, 10, 11, 12, 13, 14, 15);
}

__global__ __launch_bounds__(256)
void ltc_rk4_wmma_kernel(const float* __restrict__ inputs,
                         const float* __restrict__ state,
                         const float* __restrict__ W,
                         const float* __restrict__ bias,
                         float* __restrict__ out) {
  // Ws^T source, stored row-major [unit][k] in f16 : 32 KB
  __shared__ _Float16 sWs[UNITS * UNITS];
  // double-buffered activation staging tile [row][k] f16 : 2 * 16 KB
  __shared__ _Float16 sA[2][MTILE * UNITS];

  const int tid   = threadIdx.x;
  const int lane  = tid & 31;
  const int wave  = tid >> 5;
  const int mb    = wave & 3;   // M band (4 bands of 16 rows)
  const int nh    = wave >> 2;  // N half (2 halves of 64 units)
  const int rb    = mb * 16;    // local row base
  const int l16   = lane & 15;
  const int lhalf = lane >> 4;
  const int m0    = blockIdx.x * MTILE;
  const int ncol0 = nh * 64;

  // ---- cooperative stage of Ws (cols 64..191 of W) into LDS as f16 ----
  #pragma unroll
  for (int it = 0; it < 32; ++it) {
    int idx  = (it * 256 + tid) * 2;   // pair index over 128*128 elements
    int unit = idx >> 7;
    int k    = idx & 127;
    const float* wp = W + unit * WPITCH + IDIM + k;
    v2h p;
    p.x = (_Float16)wp[0];
    p.y = (_Float16)wp[1];
    *(v2h*)(&sWs[idx]) = p;
  }

  // ---- load state tile into registers (C/D lane layout) ----
  v8f sreg[4];
  #pragma unroll
  for (int nt = 0; nt < 4; ++nt) {
    #pragma unroll
    for (int e = 0; e < 8; ++e) {
      int r = m0 + rb + e + 8 * lhalf;
      int c = ncol0 + nt * 16 + l16;
      sreg[nt][e] = state[(size_t)r * UNITS + c];
    }
  }

  // ---- x_proj = inputs @ Wx^T + b   (accumulated into xp registers) ----
  v8f xp[4];
  #pragma unroll
  for (int nt = 0; nt < 4; ++nt) {
    float bb = bias[ncol0 + nt * 16 + l16];
    #pragma unroll
    for (int e = 0; e < 8; ++e) xp[nt][e] = bb;
  }
  #pragma unroll
  for (int kt = 0; kt < 2; ++kt) {
    // A fragment straight from global inputs (row-major, contiguous per lane)
    const float* pa = inputs + (size_t)(m0 + rb + l16) * IDIM + kt * 32 + lhalf * 16;
    v16h af;
    #pragma unroll
    for (int e = 0; e < 8; ++e) {
      af[e]     = (_Float16)pa[e];        // K = base .. base+7
      af[e + 8] = (_Float16)pa[e + 8];    // K = base+8 .. base+15
    }
    #pragma unroll
    for (int nt = 0; nt < 4; ++nt) {
      // B fragment of Wx^T straight from global W (used once; no LDS staging)
      const float* pw = W + (size_t)(ncol0 + nt * 16 + l16) * WPITCH + kt * 32 + lhalf * 16;
      v16h bf;
      #pragma unroll
      for (int e = 0; e < 16; ++e) bf[e] = (_Float16)pw[e];
      xp[nt] = wmma_f16(af, bf, xp[nt]);
    }
  }

  // ---- stage initial A = f16(state) into buffer 0 ----
  #pragma unroll
  for (int nt = 0; nt < 4; ++nt) {
    #pragma unroll
    for (int e = 0; e < 8; ++e)
      sA[0][(rb + e + 8 * lhalf) * UNITS + ncol0 + nt * 16 + l16] =
          (_Float16)sreg[nt][e];
  }

  __syncthreads();

  // ---- Ws^T B fragments: resident in VGPRs for all 24 GEMMs (128 VGPRs) ----
  v16h Bw[4][4];
  #pragma unroll
  for (int kt = 0; kt < 4; ++kt) {
    #pragma unroll
    for (int nt = 0; nt < 4; ++nt) {
      const _Float16* p = sWs + (ncol0 + nt * 16 + l16) * UNITS + kt * 32 + lhalf * 16;
      Bw[kt][nt] = cat16(*(const v8h*)p, *(const v8h*)(p + 8));
    }
  }

  v8f ksum[4];
  #pragma unroll
  for (int nt = 0; nt < 4; ++nt)
    #pragma unroll
    for (int e = 0; e < 8; ++e) ksum[nt][e] = 0.0f;

  auto do_stage = [&](const _Float16* rdA, _Float16* wrA,
                      float alphaNext, float w, bool finalStage) {
    // A fragments of current evaluation input, in WMMA A layout
    v16h Af[4];
    #pragma unroll
    for (int kt = 0; kt < 4; ++kt) {
      const _Float16* p = rdA + (rb + l16) * UNITS + kt * 32 + lhalf * 16;
      Af[kt] = cat16(*(const v8h*)p, *(const v8h*)(p + 8));
    }
    v8f acc[4];
    #pragma unroll
    for (int nt = 0; nt < 4; ++nt) acc[nt] = xp[nt];
    #pragma unroll
    for (int kt = 0; kt < 4; ++kt) {
      #pragma unroll
      for (int nt = 0; nt < 4; ++nt)
        acc[nt] = wmma_f16(Af[kt], Bw[kt][nt], acc[nt]);
    }
    // k_i = dt * tanh(.) ; accumulate RK4 weighted sum
    #pragma unroll
    for (int nt = 0; nt < 4; ++nt) {
      #pragma unroll
      for (int e = 0; e < 8; ++e) {
        float kv = DT * fast_tanh(acc[nt][e]);
        ksum[nt][e] += w * kv;
        acc[nt][e] = kv;
      }
    }
    if (finalStage) {
      #pragma unroll
      for (int nt = 0; nt < 4; ++nt) {
        #pragma unroll
        for (int e = 0; e < 8; ++e) {
          sreg[nt][e] += ksum[nt][e] * (1.0f / 6.0f);
          ksum[nt][e] = 0.0f;
        }
      }
      #pragma unroll
      for (int nt = 0; nt < 4; ++nt) {
        #pragma unroll
        for (int e = 0; e < 8; ++e)
          wrA[(rb + e + 8 * lhalf) * UNITS + ncol0 + nt * 16 + l16] =
              (_Float16)sreg[nt][e];
      }
    } else {
      #pragma unroll
      for (int nt = 0; nt < 4; ++nt) {
        #pragma unroll
        for (int e = 0; e < 8; ++e)
          wrA[(rb + e + 8 * lhalf) * UNITS + ncol0 + nt * 16 + l16] =
              (_Float16)(sreg[nt][e] + alphaNext * acc[nt][e]);
      }
    }
    __syncthreads();
  };

  for (int u = 0; u < UNFOLDS; ++u) {
    do_stage(sA[0], sA[1], 0.5f, 1.0f, false);  // k1 -> input s + 0.5 k1
    do_stage(sA[1], sA[0], 0.5f, 2.0f, false);  // k2 -> input s + 0.5 k2
    do_stage(sA[0], sA[1], 1.0f, 2.0f, false);  // k3 -> input s + k3
    do_stage(sA[1], sA[0], 0.0f, 1.0f, true);   // k4 -> s update, restage s
  }

  // ---- write (state, state) ----
  float* out1 = out + (size_t)BATCH * UNITS;
  #pragma unroll
  for (int nt = 0; nt < 4; ++nt) {
    #pragma unroll
    for (int e = 0; e < 8; ++e) {
      int r = m0 + rb + e + 8 * lhalf;
      int c = ncol0 + nt * 16 + l16;
      float v = sreg[nt][e];
      out[(size_t)r * UNITS + c]  = v;
      out1[(size_t)r * UNITS + c] = v;
    }
  }
}

extern "C" void kernel_launch(void* const* d_in, const int* in_sizes, int n_in,
                              void* d_out, int out_size, void* d_ws, size_t ws_size,
                              hipStream_t stream) {
  (void)in_sizes; (void)n_in; (void)out_size; (void)d_ws; (void)ws_size;
  const float* inputs = (const float*)d_in[0];
  const float* state  = (const float*)d_in[1];
  const float* W      = (const float*)d_in[2];
  const float* bias   = (const float*)d_in[3];
  float* out          = (float*)d_out;

  dim3 grid(BATCH / MTILE);   // 1024 workgroups
  dim3 block(256);            // 8 wave32 waves
  ltc_rk4_wmma_kernel<<<grid, block, 0, stream>>>(inputs, state, W, bias, out);
}
